// AdapterNequip_61074434949145
// MI455X (gfx1250) — matrix operand order repaired and checked by reading, not compiled, not run on hardware
//
#include <hip/hip_runtime.h>
#include <cmath>

// ------------------------------ problem constants ------------------------------
constexpr int    kN    = 10000;
constexpr int    kE    = 160000;
constexpr int    kA    = 64;      // node attr dim
constexpr int    kF    = 128;     // feature dim
constexpr int    kFD   = 32;      // adapter hidden
constexpr int    kH    = 64;      // edge MLP hidden
constexpr int    kEC   = 16000;   // edge chunk (multiple of 16)
constexpr int    kKSC  = kF * kA; // 8192, K of the self-connection einsum
constexpr size_t kNPF  = (size_t)kN * kF;   // one N x F plane
constexpr size_t kNPFD = (size_t)kN * kFD;  // one N x FD plane

// scale constants (lin_s/lin_v divide by sqrt(fan-in))
#define RSQ_F    0.08838834764831845f   // 1/sqrt(128)
#define RSQ_FD   0.17677669529663687f   // 1/sqrt(32)
#define RSQ_NB   0.35355339059327373f   // 1/sqrt(8)
#define RSQ_H    0.125f                 // 1/sqrt(64)
#define RSQ_A    0.125f                 // 1/sqrt(64)
#define RSQ_KSC  0.011048543456039806f  // 1/sqrt(8192)
#define INV_AVG  0.25f                  // 1/sqrt(16)
#define INV_SQ3  0.5773502691896258f
#define INV_SQ2  0.7071067811865475f

typedef __attribute__((ext_vector_type(16))) __bf16 bf16x16;
typedef __attribute__((ext_vector_type(8)))  float  f32x8;

__device__ __forceinline__ float silu_f(float x) { return x / (1.f + __expf(-x)); }

__device__ __forceinline__ void atomic_add_f32(float* p, float v) {
  __hip_atomic_fetch_add(p, v, __ATOMIC_RELAXED, __HIP_MEMORY_SCOPE_AGENT);
}

__device__ __forceinline__ void pack8(bf16x16& a, int base, float4 lo, float4 hi) {
  a[base + 0] = (__bf16)lo.x; a[base + 1] = (__bf16)lo.y;
  a[base + 2] = (__bf16)lo.z; a[base + 3] = (__bf16)lo.w;
  a[base + 4] = (__bf16)hi.x; a[base + 5] = (__bf16)hi.y;
  a[base + 6] = (__bf16)hi.z; a[base + 7] = (__bf16)hi.w;
}

// ------------------------------------------------------------------------------
// Weight prep: fp32 (Ksrc x Nc, row-major) -> bf16 B^T (Nc x Kpad), zero padded.
// K-major rows make each WMMA B-fragment a single contiguous 32B load per lane.
// ------------------------------------------------------------------------------
__global__ void k_prep_w(const float* __restrict__ W, __bf16* __restrict__ Bt,
                         int Ksrc, int Kpad, int Nc) {
  int t = blockIdx.x * blockDim.x + threadIdx.x;
  if (t >= Nc * Kpad) return;
  int col = t / Kpad;
  int k   = t - col * Kpad;
  float v = (k < Ksrc) ? W[(size_t)k * Nc + col] : 0.f;
  Bt[t] = (__bf16)v;
}

// ------------------------------------------------------------------------------
// Generic bf16 WMMA GEMM: C(MxNc) = act(alpha * A(MxK) @ B(KxNc) + bias)
//   A: fp32 row-major (lda), converted to bf16 on the fly
//   Bt: bf16 B^T (Nc x K, K-major)
// One wave -> 16x32 output tile, two f32x8 accumulators, 2 WMMA per K-step.
// M % 16 == 0, Nc % 32 == 0, K % 32 == 0 guaranteed by the caller.
// ------------------------------------------------------------------------------
__global__ __launch_bounds__(256) void k_gemm_bf16(
    const float* __restrict__ A, int lda,
    const __bf16* __restrict__ Bt,
    float* __restrict__ C, int ldc,
    int M, int Nc, int K, float alpha,
    const float* __restrict__ bias, int act, int accum) {
  int wave  = blockIdx.x * (blockDim.x >> 5) + (threadIdx.x >> 5);
  int tN2   = Nc >> 5;
  int tiles = (M >> 4) * tN2;
  if (wave >= tiles) return;           // wave-uniform: EXEC stays all-ones
  int tm = wave / tN2, tn = wave % tN2;
  int lane = threadIdx.x & 31;
  int m    = lane & 15;
  int half = lane >> 4;

  const float*  Arow = A + (size_t)(tm * 16 + m) * lda;
  int colA = tn * 32 + m;
  int colB = colA + 16;
  const __bf16* B0 = Bt + (size_t)colA * K + half * 16;
  const __bf16* B1 = Bt + (size_t)colB * K + half * 16;

  f32x8 c0 = {0.f,0.f,0.f,0.f,0.f,0.f,0.f,0.f};
  f32x8 c1 = {0.f,0.f,0.f,0.f,0.f,0.f,0.f,0.f};

  for (int kb = 0; kb < K; kb += 32) {
    bf16x16 a;
    pack8(a, 0, *(const float4*)(Arow + kb + half * 16),
                *(const float4*)(Arow + kb + half * 16 + 4));
    pack8(a, 8, *(const float4*)(Arow + kb + half * 16 + 8),
                *(const float4*)(Arow + kb + half * 16 + 12));
    bf16x16 b0 = *(const bf16x16*)(B0 + kb);
    bf16x16 b1 = *(const bf16x16*)(B1 + kb);
    c0 = __builtin_amdgcn_wmma_f32_16x16x32_bf16(false, a, false, b0, (short)0, c0, false, false);
    c1 = __builtin_amdgcn_wmma_f32_16x16x32_bf16(false, a, false, b1, (short)0, c1, false, false);
    if (kb + 32 < K) {                 // next-chunk prefetch -> global_prefetch_b8
      __builtin_prefetch(B0 + kb + 32, 0, 1);
      __builtin_prefetch(B1 + kb + 32, 0, 1);
    }
  }

  float biasA = bias ? bias[colA] : 0.f;
  float biasB = bias ? bias[colB] : 0.f;
#pragma unroll
  for (int r = 0; r < 8; ++r) {
    int row = tm * 16 + r + 8 * half;  // C/D layout: M = r + 8*(lane>>4)
    float x0 = c0[r] * alpha + biasA;
    float x1 = c1[r] * alpha + biasB;
    if (act) { x0 = silu_f(x0); x1 = silu_f(x1); }
    size_t o0 = (size_t)row * ldc + colA;
    size_t o1 = (size_t)row * ldc + colB;
    if (accum) { C[o0] += x0; C[o1] += x1; }
    else       { C[o0]  = x0; C[o1]  = x1; }
  }
}

// ------------------------------------------------------------------------------
// Self-connection einsum as a WMMA GEMM with on-the-fly A:
//   C[n,o] += alpha * sum_{f,a} S[n,f]*Attr[n,a]*W[f,a,o],   K = F*A = 8192.
// A 32-wide K chunk (k = f*64+a) lives inside a single f, so each lane's A
// fragment is S[n,f] * (contiguous attrs slice) - no N x 8192 materialization.
// ------------------------------------------------------------------------------
__global__ __launch_bounds__(256) void k_gemm_sc(
    const float* __restrict__ S,     // N x F
    const float* __restrict__ Attr,  // N x 64
    const __bf16* __restrict__ Bt,   // Nc x 8192
    float* __restrict__ C, int ldc, int M, int Nc, float alpha) {
  int wave  = blockIdx.x * (blockDim.x >> 5) + (threadIdx.x >> 5);
  int tN2   = Nc >> 5;
  int tiles = (M >> 4) * tN2;
  if (wave >= tiles) return;
  int tm = wave / tN2, tn = wave % tN2;
  int lane = threadIdx.x & 31;
  int m    = lane & 15;
  int half = lane >> 4;

  const float* Srow = S    + (size_t)(tm * 16 + m) * kF;
  const float* Trow = Attr + (size_t)(tm * 16 + m) * kA;
  int colA = tn * 32 + m;
  int colB = colA + 16;
  const __bf16* B0 = Bt + (size_t)colA * kKSC + half * 16;
  const __bf16* B1 = Bt + (size_t)colB * kKSC + half * 16;

  f32x8 c0 = {0.f,0.f,0.f,0.f,0.f,0.f,0.f,0.f};
  f32x8 c1 = {0.f,0.f,0.f,0.f,0.f,0.f,0.f,0.f};

  for (int kb = 0; kb < kKSC; kb += 32) {
    int f     = kb >> 6;        // A = 64 -> f fixed within a 32-chunk
    int abase = kb & 63;        // 0 or 32
    float sA  = Srow[f];
    const float* ap = Trow + abase;
    float4 l0 = *(const float4*)(ap + half * 16);
    float4 h0 = *(const float4*)(ap + half * 16 + 4);
    float4 l1 = *(const float4*)(ap + half * 16 + 8);
    float4 h1 = *(const float4*)(ap + half * 16 + 12);
    l0.x *= sA; l0.y *= sA; l0.z *= sA; l0.w *= sA;
    h0.x *= sA; h0.y *= sA; h0.z *= sA; h0.w *= sA;
    l1.x *= sA; l1.y *= sA; l1.z *= sA; l1.w *= sA;
    h1.x *= sA; h1.y *= sA; h1.z *= sA; h1.w *= sA;
    bf16x16 a;
    pack8(a, 0, l0, h0);
    pack8(a, 8, l1, h1);
    bf16x16 b0 = *(const bf16x16*)(B0 + kb);
    bf16x16 b1 = *(const bf16x16*)(B1 + kb);
    c0 = __builtin_amdgcn_wmma_f32_16x16x32_bf16(false, a, false, b0, (short)0, c0, false, false);
    c1 = __builtin_amdgcn_wmma_f32_16x16x32_bf16(false, a, false, b1, (short)0, c1, false, false);
    if (kb + 32 < kKSC) {
      __builtin_prefetch(B0 + kb + 32, 0, 1);
      __builtin_prefetch(B1 + kb + 32, 0, 1);
    }
  }
#pragma unroll
  for (int r = 0; r < 8; ++r) {
    int row = tm * 16 + r + 8 * half;
    C[(size_t)row * ldc + colA] += c0[r] * alpha;
    C[(size_t)row * ldc + colB] += c1[r] * alpha;
  }
}

// ------------------------------------------------------------------------------
// Edge geometry: y1 (E x 3) and zero-padded Bessel edge features (E x 32).
// ------------------------------------------------------------------------------
__global__ void k_edge_geom(const int* __restrict__ ei, const float* __restrict__ coord,
                            float* __restrict__ y1, float* __restrict__ ef) {
  int e = blockIdx.x * blockDim.x + threadIdx.x;
  if (e >= kE) return;
  int s = ei[e], r = ei[kE + e];
  float dx = coord[s * 3 + 0] - coord[r * 3 + 0];
  float dy = coord[s * 3 + 1] - coord[r * 3 + 1];
  float dz = coord[s * 3 + 2] - coord[r * 3 + 2];
  float len  = sqrtf(dx * dx + dy * dy + dz * dz + 1e-18f);
  float invl = 1.f / len;
  const float SQ3 = 1.7320508075688772f;
  y1[e * 3 + 0] = SQ3 * dx * invl;
  y1[e * 3 + 1] = SQ3 * dy * invl;
  y1[e * 3 + 2] = SQ3 * dz * invl;
  float x  = len * 0.2f;                 // len / R_MAX
  float x2 = x * x;
  float x6 = x2 * x2 * x2;
  float env = (x < 1.f) ? (1.f - 28.f * x6 + 48.f * x6 * x - 21.f * x6 * x2) : 0.f;
  float pref = 0.6324555320336759f * invl * env;   // sqrt(2/5)/len * env
  const float w0 = 0.6283185307179586f;            // pi/5
#pragma unroll
  for (int n = 1; n <= 8; ++n)
    ef[(size_t)e * 32 + n - 1] = pref * __sinf((float)n * w0 * len);
#pragma unroll
  for (int k = 8; k < 32; ++k) ef[(size_t)e * 32 + k] = 0.f;
}

// ------------------------------------------------------------------------------
// Per-edge messages + segment-sum scatter (native relaxed f32 atomics).
// v planes are stored c-major: v[c*kNPF + n*F + f].
// ------------------------------------------------------------------------------
__global__ void k_messages(const int* __restrict__ snd, const int* __restrict__ rcv,
                           const float* __restrict__ s1, const float* __restrict__ v1,
                           const float* __restrict__ y1, const float* __restrict__ w,
                           float* __restrict__ aggs, float* __restrict__ aggv,
                           int nP, int nE) {
  int t = blockIdx.x * blockDim.x + threadIdx.x;
  if (t >= nE * kF) return;
  int e = t >> 7;
  int f = t & (kF - 1);
  int src = snd[e], dst = rcv[e];
  float sj = s1[(size_t)src * kF + f];
  const float* we = w + (size_t)e * nP * kF;
  float w0v = we[f];
  float w1v = we[kF + f];
  float ya = y1[e * 3 + 0], yb = y1[e * 3 + 1], yc = y1[e * 3 + 2];
  float ms  = w0v * sj;
  float mv0 = w1v * sj * ya;
  float mv1 = w1v * sj * yb;
  float mv2 = w1v * sj * yc;
  if (nP == 5) {
    size_t o = (size_t)src * kF + f;
    float vj0 = v1[o], vj1 = v1[kNPF + o], vj2 = v1[2 * kNPF + o];
    float w2v = we[2 * kF + f], w3v = we[3 * kF + f], w4v = we[4 * kF + f];
    ms  += w2v * (vj0 * ya + vj1 * yb + vj2 * yc) * INV_SQ3;
    mv0 += w3v * vj0 + w4v * (vj1 * yc - vj2 * yb) * INV_SQ2;
    mv1 += w3v * vj1 + w4v * (vj2 * ya - vj0 * yc) * INV_SQ2;
    mv2 += w3v * vj2 + w4v * (vj0 * yb - vj1 * ya) * INV_SQ2;
  }
  size_t od = (size_t)dst * kF + f;
  atomic_add_f32(&aggs[od],             ms  * INV_AVG);
  atomic_add_f32(&aggv[od],             mv0 * INV_AVG);
  atomic_add_f32(&aggv[kNPF + od],      mv1 * INV_AVG);
  atomic_add_f32(&aggv[2 * kNPF + od],  mv2 * INV_AVG);
}

// ------------------------------ batchnorm pieces ------------------------------
__global__ void k_bn_stats_s(const float* __restrict__ X, int M,
                             float* __restrict__ mu, float* __restrict__ var) {
  __shared__ float sh[256], sh2[256];
  int f = blockIdx.x;
  float a = 0.f, b = 0.f;
  for (int n = threadIdx.x; n < M; n += blockDim.x) {
    float x = X[(size_t)n * kF + f];
    a += x; b += x * x;
  }
  sh[threadIdx.x] = a; sh2[threadIdx.x] = b; __syncthreads();
  for (int st = 128; st > 0; st >>= 1) {
    if ((int)threadIdx.x < st) { sh[threadIdx.x] += sh[threadIdx.x + st]; sh2[threadIdx.x] += sh2[threadIdx.x + st]; }
    __syncthreads();
  }
  if (threadIdx.x == 0) {
    float m = sh[0] / (float)M;
    mu[f] = m; var[f] = sh2[0] / (float)M - m * m;
  }
}

__global__ void k_bn_stats_v(const float* __restrict__ V, int M, float* __restrict__ vn) {
  __shared__ float sh[256];
  int f = blockIdx.x;
  float a = 0.f;
  for (int n = threadIdx.x; n < M; n += blockDim.x) {
    size_t o = (size_t)n * kF + f;
    float x0 = V[o], x1 = V[kNPF + o], x2 = V[2 * kNPF + o];
    a += x0 * x0 + x1 * x1 + x2 * x2;
  }
  sh[threadIdx.x] = a; __syncthreads();
  for (int st = 128; st > 0; st >>= 1) {
    if ((int)threadIdx.x < st) sh[threadIdx.x] += sh[threadIdx.x + st];
    __syncthreads();
  }
  if (threadIdx.x == 0) vn[f] = sh[0] / (float)(M * 3);
}

__global__ void k_bn_apply(const float* __restrict__ su, const float* __restrict__ vu,
                           const float* __restrict__ mu, const float* __restrict__ var,
                           const float* __restrict__ vn,
                           const float* __restrict__ ws, const float* __restrict__ bs,
                           const float* __restrict__ wv,
                           float* __restrict__ outs, float* __restrict__ outv) {
  int t = blockIdx.x * blockDim.x + threadIdx.x;
  if (t >= (int)kNPF) return;
  int f = t & (kF - 1);
  float ks = ws[f] * rsqrtf(var[f] + 1e-5f);
  outs[t] = (su[t] - mu[f]) * ks + bs[f];
  float kv = wv[f] * rsqrtf(vn[f] + 1e-5f);
#pragma unroll
  for (int c = 0; c < 3; ++c) {
    size_t o = (size_t)c * kNPF + t;
    outv[o] = vu[o] * kv;
  }
}

// gate/split epilogue of conv_layer
__global__ void k_combine(const float* __restrict__ sg, const float* __restrict__ v2,
                          const float* __restrict__ s, const float* __restrict__ v,
                          float* __restrict__ cs, float* __restrict__ cv, int hv) {
  int t = blockIdx.x * blockDim.x + threadIdx.x;
  if (t >= (int)kNPF) return;
  int n = t >> 7, f = t & (kF - 1);
  float ns = sg[(size_t)n * 256 + f];
  float g  = sg[(size_t)n * 256 + kF + f];
  float os = silu_f(ns);
  float sig = silu_f(g);
  if (hv) os += s[t];
  cs[t] = os;
#pragma unroll
  for (int c = 0; c < 3; ++c) {
    size_t o = (size_t)c * kNPF + t;
    float ov = v2[o] * sig;
    if (hv) ov += v[o];
    cv[o] = ov;
  }
}

// s = bn(cs) + sb*ab_s + sa*aa_s ; v likewise (scalars read from device params)
__global__ void k_update(float* __restrict__ s, float* __restrict__ v,
                         const float* __restrict__ cs, const float* __restrict__ cv,
                         const float* __restrict__ mu, const float* __restrict__ var,
                         const float* __restrict__ vn,
                         const float* __restrict__ bw, const float* __restrict__ bb,
                         const float* __restrict__ bwv,
                         const float* __restrict__ ab_s, const float* __restrict__ ab_v,
                         const float* __restrict__ aa_s, const float* __restrict__ aa_v,
                         const float* __restrict__ sb_p, const float* __restrict__ sa_p) {
  int t = blockIdx.x * blockDim.x + threadIdx.x;
  if (t >= (int)kNPF) return;
  int f = t & (kF - 1);
  float sb = *sb_p, sa = *sa_p;
  float ks = bw[f] * rsqrtf(var[f] + 1e-5f);
  s[t] = (cs[t] - mu[f]) * ks + bb[f] + sb * ab_s[t] + sa * aa_s[t];
  float kv = bwv[f] * rsqrtf(vn[f] + 1e-5f);
#pragma unroll
  for (int c = 0; c < 3; ++c) {
    size_t o = (size_t)c * kNPF + t;
    v[o] = cv[o] * kv + sb * ab_v[o] + sa * aa_v[o];
  }
}

// final concat: out[n, 0:128] = s ; out[n, 128 + f*3 + c] = v[c][n,f]
__global__ void k_final(const float* __restrict__ s, const float* __restrict__ v,
                        float* __restrict__ out) {
  int t = blockIdx.x * blockDim.x + threadIdx.x;
  if (t >= (int)kNPF) return;
  int n = t >> 7, f = t & (kF - 1);
  out[(size_t)n * 512 + f] = s[t];
#pragma unroll
  for (int c = 0; c < 3; ++c)
    out[(size_t)n * 512 + kF + f * 3 + c] = v[(size_t)c * kNPF + t];
}

// ==============================================================================
extern "C" void kernel_launch(void* const* d_in, const int* in_sizes, int n_in,
                              void* d_out, int out_size, void* d_ws, size_t ws_size,
                              hipStream_t stream) {
  (void)in_sizes; (void)n_in; (void)out_size; (void)ws_size;

  const float* node_attrs = (const float*)d_in[0];
  const float* coords     = (const float*)d_in[1];
  const int*   eidx       = (const int*)d_in[2];

  // ---- parameter pytree (sorted dict keys, JAX flatten order) ----
  int pi = 3;
  auto P = [&](void) { return (const float*)d_in[pi++]; };
  const float* W_embed = P();
  struct AdaP { const float *bn_b,*bn_w,*bn_wv,*ld_s,*ld_v,*lu_s,*lu_v; };
  struct LayP {
    AdaP after, before;
    const float *bn_b,*bn_w,*bn_wv,*lin1_s,*lin1_v,*lin2_s,*lin2_v;
    const float *W1,*W2,*W3,*b1,*b2,*s_after,*s_before,*sc_s,*sc_v;
  };
  LayP lp[3];
  for (int i = 0; i < 3; ++i) {
    bool hv = (i > 0);
    auto rdA = [&](bool av) {
      AdaP a{}; a.bn_b = P(); a.bn_w = P(); a.bn_wv = P();
      a.ld_s = P(); if (av) a.ld_v = P();
      a.lu_s = P(); if (av) a.lu_v = P();
      return a;
    };
    lp[i].after  = rdA(true);
    lp[i].before = rdA(hv);
    lp[i].bn_b = P(); lp[i].bn_w = P(); lp[i].bn_wv = P();
    lp[i].lin1_s = P(); lp[i].lin1_v = hv ? P() : nullptr;
    lp[i].lin2_s = P(); lp[i].lin2_v = P();
    lp[i].W1 = P(); lp[i].W2 = P(); lp[i].W3 = P(); lp[i].b1 = P(); lp[i].b2 = P();
    lp[i].s_after = P(); lp[i].s_before = P();
    lp[i].sc_s = P(); lp[i].sc_v = hv ? P() : nullptr;
  }

  // ---- workspace bump allocator (256B aligned) ----
  char* wsp = (char*)d_ws;
  auto bump = [&](size_t bytes) -> void* {
    void* r = wsp; wsp += (bytes + 255) & ~(size_t)255; return r;
  };
  float* s    = (float*)bump(kNPF * 4);
  float* v    = (float*)bump(3 * kNPF * 4);
  float* cs   = (float*)bump(kNPF * 4);
  float* cv   = (float*)bump(3 * kNPF * 4);
  float* ab_s = (float*)bump(kNPF * 4);
  float* ab_v = (float*)bump(3 * kNPF * 4);
  float* aa_s = (float*)bump(kNPF * 4);
  float* aa_v = (float*)bump(3 * kNPF * 4);
  float* s1b  = (float*)bump(kNPF * 4);
  float* v1b  = (float*)bump(3 * kNPF * 4);
  float* aggs = (float*)bump(kNPF * 4);
  float* aggv = (float*)bump(3 * kNPF * 4);
  float* sg   = (float*)bump((size_t)kN * 2 * kF * 4);
  float* v2b  = (float*)bump(3 * kNPF * 4);
  float* sd   = (float*)bump(kNPFD * 4);
  float* vd   = (float*)bump(3 * kNPFD * 4);
  float* su   = (float*)bump(kNPF * 4);
  float* vu   = (float*)bump(3 * kNPF * 4);
  float* y1   = (float*)bump((size_t)kE * 3 * 4);
  float* ef   = (float*)bump((size_t)kE * 32 * 4);
  float* h1   = (float*)bump((size_t)kEC * kH * 4);
  float* h2   = (float*)bump((size_t)kEC * kH * 4);
  float* wbuf = (float*)bump((size_t)kEC * 5 * kF * 4);
  float* mu   = (float*)bump(kF * 4);
  float* var  = (float*)bump(kF * 4);
  float* vn   = (float*)bump(kF * 4);

  // ---- bf16 transposed weight prep ----
  auto prep = [&](const float* W, int Ksrc, int Kpad, int Nc) -> const __bf16* {
    __bf16* bt = (__bf16*)bump((size_t)Kpad * Nc * 2);
    int tot = Kpad * Nc;
    k_prep_w<<<(tot + 255) / 256, 256, 0, stream>>>(W, bt, Ksrc, Kpad, Nc);
    return bt;
  };
  const __bf16* bt_embed = prep(W_embed, kA, kA, kF);
  struct LayW {
    const __bf16 *lin1_s,*lin1_v,*lin2_s,*lin2_v,*W1,*W2,*W3,*sc_s,*sc_v;
    const __bf16 *af_ld_s,*af_ld_v,*af_lu_s,*af_lu_v;
    const __bf16 *bf_ld_s,*bf_ld_v,*bf_lu_s,*bf_lu_v;
  };
  LayW lw[3];
  for (int i = 0; i < 3; ++i) {
    bool hv = (i > 0); int nP = hv ? 5 : 2;
    lw[i].lin1_s = prep(lp[i].lin1_s, kF, kF, kF);
    lw[i].lin1_v = hv ? prep(lp[i].lin1_v, kF, kF, kF) : nullptr;
    lw[i].lin2_s = prep(lp[i].lin2_s, kF, kF, 2 * kF);
    lw[i].lin2_v = prep(lp[i].lin2_v, kF, kF, kF);
    lw[i].W1 = prep(lp[i].W1, 8, 32, kH);        // zero-padded K: 8 -> 32
    lw[i].W2 = prep(lp[i].W2, kH, kH, kH);
    lw[i].W3 = prep(lp[i].W3, kH, kH, nP * kF);
    lw[i].sc_s = prep(lp[i].sc_s, kKSC, kKSC, 2 * kF);
    lw[i].sc_v = hv ? prep(lp[i].sc_v, kKSC, kKSC, kF) : nullptr;
    lw[i].af_ld_s = prep(lp[i].after.ld_s, kF, kF, kFD);
    lw[i].af_ld_v = prep(lp[i].after.ld_v, kF, kF, kFD);
    lw[i].af_lu_s = prep(lp[i].after.lu_s, kFD, kFD, kF);
    lw[i].af_lu_v = prep(lp[i].after.lu_v, kFD, kFD, kF);
    lw[i].bf_ld_s = prep(lp[i].before.ld_s, kF, kF, kFD);
    lw[i].bf_ld_v = hv ? prep(lp[i].before.ld_v, kF, kF, kFD) : nullptr;
    lw[i].bf_lu_s = prep(lp[i].before.lu_s, kFD, kFD, kF);
    lw[i].bf_lu_v = hv ? prep(lp[i].before.lu_v, kFD, kFD, kF) : nullptr;
  }

  auto gemm = [&](const float* A, int lda, const __bf16* Bt, float* C, int ldc,
                  int M, int Nc, int K, float alpha, const float* bias,
                  int act, int accum) {
    int tiles  = (M / 16) * (Nc / 32);
    int blocks = (tiles + 7) / 8;
    k_gemm_bf16<<<blocks, 256, 0, stream>>>(A, lda, Bt, C, ldc, M, Nc, K,
                                            alpha, bias, act, accum);
  };
  auto gemm_sc = [&](const float* S, const __bf16* Bt, float* C, int ldc, int Nc) {
    int tiles  = (kN / 16) * (Nc / 32);
    int blocks = (tiles + 7) / 8;
    k_gemm_sc<<<blocks, 256, 0, stream>>>(S, node_attrs, Bt, C, ldc, kN, Nc, RSQ_KSC);
  };

  const int ewBlocks = ((int)kNPF + 255) / 256;

  // adapter: out_s/out_v = batchnorm( lin(silu(lin(s))), lin(lin(v)) )
  auto run_adapter = [&](const float* in_s, const float* in_v, bool hv,
                         const __bf16* ld_s, const __bf16* ld_v,
                         const __bf16* lu_s, const __bf16* lu_v,
                         const AdaP& ap, float* out_s, float* out_v) {
    gemm(in_s, kF, ld_s, sd, kFD, kN, kFD, kF, RSQ_F, nullptr, 1, 0);   // silu
    gemm(sd, kFD, lu_s, su, kF, kN, kF, kFD, RSQ_FD, nullptr, 0, 0);
    if (hv) {
      for (int c = 0; c < 3; ++c)
        gemm(in_v + (size_t)c * kNPF, kF, ld_v, vd + (size_t)c * kNPFD, kFD,
             kN, kFD, kF, RSQ_F, nullptr, 0, 0);
      for (int c = 0; c < 3; ++c)
        gemm(vd + (size_t)c * kNPFD, kFD, lu_v, vu + (size_t)c * kNPF, kF,
             kN, kF, kFD, RSQ_FD, nullptr, 0, 0);
    } else {
      hipMemsetAsync(vu, 0, 3 * kNPF * 4, stream);
    }
    k_bn_stats_s<<<kF, 256, 0, stream>>>(su, kN, mu, var);
    k_bn_stats_v<<<kF, 256, 0, stream>>>(vu, kN, vn);
    k_bn_apply<<<ewBlocks, 256, 0, stream>>>(su, vu, mu, var, vn,
                                             ap.bn_w, ap.bn_b, ap.bn_wv,
                                             out_s, out_v);
  };

  // ------------------------------ forward pass ------------------------------
  k_edge_geom<<<(kE + 255) / 256, 256, 0, stream>>>(eidx, coords, y1, ef);
  gemm(node_attrs, kA, bt_embed, s, kF, kN, kF, kA, RSQ_A, nullptr, 0, 0);
  hipMemsetAsync(v, 0, 3 * kNPF * 4, stream);

  for (int i = 0; i < 3; ++i) {
    bool hv = (i > 0); int nP = hv ? 5 : 2;

    // adapter-before on (s, v)
    run_adapter(s, v, hv, lw[i].bf_ld_s, lw[i].bf_ld_v, lw[i].bf_lu_s, lw[i].bf_lu_v,
                lp[i].before, ab_s, ab_v);

    // conv layer
    gemm(s, kF, lw[i].lin1_s, s1b, kF, kN, kF, kF, RSQ_F, nullptr, 0, 0);
    if (hv)
      for (int c = 0; c < 3; ++c)
        gemm(v + (size_t)c * kNPF, kF, lw[i].lin1_v, v1b + (size_t)c * kNPF, kF,
             kN, kF, kF, RSQ_F, nullptr, 0, 0);
    hipMemsetAsync(aggs, 0, kNPF * 4, stream);
    hipMemsetAsync(aggv, 0, 3 * kNPF * 4, stream);

    for (int e0 = 0; e0 < kE; e0 += kEC) {
      gemm(ef + (size_t)e0 * 32, 32, lw[i].W1, h1, kH, kEC, kH, 32, RSQ_NB, lp[i].b1, 1, 0);
      gemm(h1, kH, lw[i].W2, h2, kH, kEC, kH, kH, RSQ_H, lp[i].b2, 1, 0);
      gemm(h2, kH, lw[i].W3, wbuf, nP * kF, kEC, nP * kF, kH, RSQ_H, nullptr, 0, 0);
      k_messages<<<(kEC * kF + 255) / 256, 256, 0, stream>>>(
          eidx + e0, eidx + kE + e0, s1b, v1b, y1 + (size_t)e0 * 3, wbuf,
          aggs, aggv, nP, kEC);
    }

    gemm(aggs, kF, lw[i].lin2_s, sg, 2 * kF, kN, 2 * kF, kF, RSQ_F, nullptr, 0, 0);
    for (int c = 0; c < 3; ++c)
      gemm(aggv + (size_t)c * kNPF, kF, lw[i].lin2_v, v2b + (size_t)c * kNPF, kF,
           kN, kF, kF, RSQ_F, nullptr, 0, 0);

    // self-connection einsums, accumulated into sg / v2
    gemm_sc(s, lw[i].sc_s, sg, 2 * kF, 2 * kF);
    if (hv)
      for (int c = 0; c < 3; ++c)
        gemm_sc(v + (size_t)c * kNPF, lw[i].sc_v, v2b + (size_t)c * kNPF, kF, kF);

    k_combine<<<ewBlocks, 256, 0, stream>>>(sg, v2b, s, v, cs, cv, hv ? 1 : 0);

    // adapter-after on (cs, cv) — always has vectors
    run_adapter(cs, cv, true, lw[i].af_ld_s, lw[i].af_ld_v, lw[i].af_lu_s, lw[i].af_lu_v,
                lp[i].after, aa_s, aa_v);

    // main batchnorm + residual mix -> next (s, v)
    k_bn_stats_s<<<kF, 256, 0, stream>>>(cs, kN, mu, var);
    k_bn_stats_v<<<kF, 256, 0, stream>>>(cv, kN, vn);
    k_update<<<ewBlocks, 256, 0, stream>>>(s, v, cs, cv, mu, var, vn,
                                           lp[i].bn_w, lp[i].bn_b, lp[i].bn_wv,
                                           ab_s, ab_v, aa_s, aa_v,
                                           lp[i].s_before, lp[i].s_after);
  }

  k_final<<<ewBlocks, 256, 0, stream>>>(s, v, (float*)d_out);
}
